// MotionTrackingLK_6227702579586
// MI455X (gfx1250) — matrix operand order repaired and checked by reading, not compiled, not run on hardware
//
#include <hip/hip_runtime.h>
#include <math.h>

#define B_    32
#define T_    512
#define SEQ_  5
#define H_    256
#define W_    256
#define WIN_  21
#define W2_   441
#define ITERS_ 5
#define KPAD  512          // K padded to 16 chunks of 32 for WMMA
#define TPG   16           // tracks per workgroup (= WMMA M)
#define INV_SCALE (256.0f / 21.0f)

typedef __attribute__((ext_vector_type(16))) _Float16 v16h;
typedef __attribute__((ext_vector_type(8)))  float    v8f;

__device__ __forceinline__ float bilinear(const float* __restrict__ img, float x, float y) {
    // Exactly mirrors the reference: clamp floor/floor+1, weights from raw x,y.
    float fx = floorf(x), fy = floorf(y);
    float x0 = fminf(fmaxf(fx,        0.f), (float)(W_ - 1));
    float x1 = fminf(fmaxf(fx + 1.f,  0.f), (float)(W_ - 1));
    float y0 = fminf(fmaxf(fy,        0.f), (float)(H_ - 1));
    float y1 = fminf(fmaxf(fy + 1.f,  0.f), (float)(H_ - 1));
    float wa = (y1 - y) * (x1 - x);
    float wb = (y1 - y) * (x  - x0);
    float wc = (y  - y0) * (x1 - x);
    float wd = (y  - y0) * (x  - x0);
    int x0i = (int)x0, x1i = (int)x1, y0i = (int)y0, y1i = (int)y1;
    float Ia = img[y0i * W_ + x0i];
    float Ib = img[y0i * W_ + x1i];
    float Ic = img[y1i * W_ + x0i];
    float Id = img[y1i * W_ + x1i];
    return wa * Ia + wb * Ib + wc * Ic + wd * Id;
}

// Weighted window reduction for all 16 tracks of the workgroup at once:
//   outS[t] = sign * sum_k p1[t][k] * p2[t][k] * w[k]
// A[t][k] = p1*p2 (f16, per ISA 16-bit A 16x32 VGPR layout), B[k][n] = w[k]
// broadcast across the 16 columns, f32 accumulate over 16 chunks of K=32.
__device__ __forceinline__ void wmma_weighted_reduce(
        const _Float16* __restrict__ p1, const _Float16* __restrict__ p2,
        const _Float16* __restrict__ wl, float* __restrict__ outS,
        float sign, int lane) {
    const int  m  = lane & 15;        // track row (WMMA M)
    const bool hi = lane >= 16;
    v8f acc = {0.f, 0.f, 0.f, 0.f, 0.f, 0.f, 0.f, 0.f};
    for (int chunk = 0; chunk < KPAD / 32; ++chunk) {
        const int base = chunk * 32;
        v16h A, Bv;
#pragma unroll
        for (int h = 0; h < 16; ++h) {
            // A-matrix 16x32 f16 layout: lanes 0-15 hold K {0..7,16..23},
            // lanes 16-31 hold K {8..15,24..31}.
            int K  = base + h + (h < 8 ? (hi ? 8 : 0) : (hi ? 16 : 8));
            A[h]   = p1[m * KPAD + K] * p2[m * KPAD + K];
            // B-matrix 32x16 f16 layout: same K order as A: lanes 0-15 hold K {0..7,16..23},
            // lanes 16-31 hold K {8..15,24..31} (column = lane%16; all columns = w).
            int Kb = base + h + (h < 8 ? (hi ? 8 : 0) : (hi ? 16 : 8));
            Bv[h]  = wl[Kb];
        }
        acc = __builtin_amdgcn_wmma_f32_16x16x32_f16(
                  false, A, false, Bv, (short)0, acc, false, false);
    }
    // D layout: VGPR j = row (j + 8*hi), column = lane%16 (all columns equal).
    if (m == 0) {
        int off = hi ? 8 : 0;
#pragma unroll
        for (int j = 0; j < 8; ++j) outS[off + j] = sign * acc[j];
    }
}

__global__ void __launch_bounds__(512, 1)
lk_track_kernel(const float* __restrict__ init_locs,
                const float* __restrict__ imgs,
                float* __restrict__ out) {
    __shared__ _Float16 s_ff[TPG][448];   // fixed window of frame p
    __shared__ _Float16 s_gx[TPG][KPAD];  // Sobel Ix (zero-padded K>=441)
    __shared__ _Float16 s_gy[TPG][KPAD];  // Sobel Iy
    __shared__ _Float16 s_it[TPG][KPAD];  // temporal diff sf - ff
    __shared__ _Float16 s_w[KPAD];        // Gaussian window weights
    __shared__ float s_sxx[TPG], s_syy[TPG], s_sxy[TPG], s_bx[TPG], s_by[TPG];

    const int tid   = threadIdx.x;
    const int lane  = tid & 31;
    const int wave  = tid >> 5;           // wave w <-> track w of this group
    const int bid   = blockIdx.x;
    const int b     = bid >> 5;           // 32 groups per batch (T/16)
    const int t0    = (bid & 31) * TPG;
    const int track = t0 + wave;

    // ---- one-time init: weights + zero K-padding of WMMA operand arrays ----
    {
        int k = tid;                       // blockDim == 512 == KPAD
        float wv = 0.f;
        if (k < W2_) {
            int r = k / WIN_, c = k % WIN_;
            float dx = (float)(c - 10), dy = (float)(r - 10);
            // 1/(sigma*sqrt(2pi)) * exp(-0.5*(r/sigma)^2), sigma = 2
            wv = 0.19947114020071635f * expf(-0.125f * (dx * dx + dy * dy));
        }
        s_w[k] = (_Float16)wv;
    }
    for (int idx = tid; idx < TPG * KPAD; idx += 512) {
        int tt = idx >> 9, kk = idx & (KPAD - 1);
        s_gx[tt][kk] = (_Float16)0.f;
        s_gy[tt][kk] = (_Float16)0.f;
        s_it[tt][kk] = (_Float16)0.f;
    }
    __syncthreads();

    // per-track offset state (wave-uniform registers): samples = GRID + offset
    float ox = init_locs[((size_t)b * T_ + track) * 2 + 0] * INV_SCALE;
    float oy = init_locs[((size_t)b * T_ + track) * 2 + 1] * INV_SCALE;

    const float KX[3][3] = {{-1.f, 0.f, 1.f}, {-2.f, 0.f, 2.f}, {-1.f, 0.f, 1.f}};
    const float KY[3][3] = {{-1.f, -2.f, -1.f}, {0.f, 0.f, 0.f}, {1.f, 2.f, 1.f}};

    for (int p = 0; p < SEQ_ - 1; ++p) {
        const float* imgA = imgs + (size_t)(b * SEQ_ + p) * (H_ * W_);
        const float* imgB = imgA + H_ * W_;

        // ---- ff: sample frame p at pair-start offsets (fixed for 5 iters) ----
        for (int k = lane; k < W2_; k += 32) {
            int r = k / WIN_, c = k % WIN_;
            float x = (-1.f + 0.1f * (float)c + ox) * 10.5f;
            float y = (-1.f + 0.1f * (float)r + oy) * 10.5f;
            float v = bilinear(imgA, x, y);
            s_ff[wave][k] = (_Float16)v;
            if (p == 0)   // first_frame output == ff of pair 0
                out[(((size_t)b * 2 + 0) * T_ + track) * W2_ + k] = v;
        }
        __syncthreads();

        // ---- Sobel (cross-correlation, SAME zero padding on 21x21) ----
        for (int k = lane; k < W2_; k += 32) {
            int r = k / WIN_, c = k % WIN_;
            float ix = 0.f, iy = 0.f;
#pragma unroll
            for (int i = 0; i < 3; ++i)
#pragma unroll
                for (int j = 0; j < 3; ++j) {
                    int rr = r + i - 1, cc = c + j - 1;
                    if (rr >= 0 && rr < WIN_ && cc >= 0 && cc < WIN_) {
                        float v = (float)s_ff[wave][rr * WIN_ + cc];
                        ix += v * KX[i][j];
                        iy += v * KY[i][j];
                    }
                }
            s_gx[wave][k] = (_Float16)ix;
            s_gy[wave][k] = (_Float16)iy;
        }
        __syncthreads();

        // ---- structure tensor: once per pair (ff fixed) via WMMA ----
        if (wave == 0)      wmma_weighted_reduce(&s_gx[0][0], &s_gx[0][0], s_w, s_sxx,  1.f, lane);
        else if (wave == 1) wmma_weighted_reduce(&s_gy[0][0], &s_gy[0][0], s_w, s_syy,  1.f, lane);
        else if (wave == 2) wmma_weighted_reduce(&s_gx[0][0], &s_gy[0][0], s_w, s_sxy,  1.f, lane);
        __syncthreads();

        // ---- LK iterations: only bx, by change per iteration ----
        for (int it = 0; it < ITERS_; ++it) {
            for (int k = lane; k < W2_; k += 32) {
                int r = k / WIN_, c = k % WIN_;
                float x = (-1.f + 0.1f * (float)c + ox) * 10.5f;
                float y = (-1.f + 0.1f * (float)r + oy) * 10.5f;
                float sf = bilinear(imgB, x, y);
                s_it[wave][k] = (_Float16)(sf - (float)s_ff[wave][k]);
            }
            __syncthreads();

            if (wave == 0)      wmma_weighted_reduce(&s_gx[0][0], &s_it[0][0], s_w, s_bx, -1.f, lane);
            else if (wave == 1) wmma_weighted_reduce(&s_gy[0][0], &s_it[0][0], s_w, s_by, -1.f, lane);
            __syncthreads();

            float sxx = s_sxx[wave], syy = s_syy[wave], sxy = s_sxy[wave];
            float bx = s_bx[wave], by = s_by[wave];
            float det_inv = 1.f / (sxx * syy - sxy * sxy + 1e-7f);
            ox += det_inv * ( syy * bx - sxy * by);
            oy += det_inv * (-sxy * bx + sxx * by);
        }
    }

    // ---- tracked output: frame SEQ-1 sampled at final offsets ----
    const float* imgF = imgs + (size_t)(b * SEQ_ + (SEQ_ - 1)) * (H_ * W_);
    for (int k = lane; k < W2_; k += 32) {
        int r = k / WIN_, c = k % WIN_;
        float x = (-1.f + 0.1f * (float)c + ox) * 10.5f;
        float y = (-1.f + 0.1f * (float)r + oy) * 10.5f;
        float v = bilinear(imgF, x, y);
        out[(((size_t)b * 2 + 1) * T_ + track) * W2_ + k] = v;
    }
}

extern "C" void kernel_launch(void* const* d_in, const int* in_sizes, int n_in,
                              void* d_out, int out_size, void* d_ws, size_t ws_size,
                              hipStream_t stream) {
    (void)in_sizes; (void)n_in; (void)out_size; (void)d_ws; (void)ws_size;
    const float* init_locs = (const float*)d_in[0];   // (B, T, 2) f32
    const float* imgs      = (const float*)d_in[1];   // (B, SEQ, H, W, 1) f32
    float* out = (float*)d_out;                        // (B, 2, T, 21, 21, 1) f32

    dim3 grid(B_ * (T_ / TPG));   // 1024 workgroups: one per 16 tracks
    dim3 block(512);              // 16 waves (wave32): one wave per track
    lk_track_kernel<<<grid, block, 0, stream>>>(init_locs, imgs, out);
}
